// Attention_81776177316478
// MI455X (gfx1250) — matrix operand order repaired and checked by reading, not compiled, not run on hardware
//
#include <hip/hip_runtime.h>
#include <hip/hip_bf16.h>

// ---------------------------------------------------------------------------
// Multi-head attention (B=4, N=2048, C=1024, H=16, D=64), fp32 in/out.
// bf16 WMMA (v_wmma_f32_16x16x32_bf16) for all four GEMM stages, flash-style
// attention (no materialized NxN score matrix).
//
// GEMM kernels: 256 threads / 8 waves, 128x128 block tile, 32x64 wave tile
// (8 WMMA per 6 fragment loads to keep the matrix pipes off the LDS-bandwidth
// ceiling). Attention: 128 threads / 4 waves, 64-query x 64-key flash blocks.
// ---------------------------------------------------------------------------

typedef __attribute__((ext_vector_type(16))) __bf16 v16bf;
typedef __attribute__((ext_vector_type(8)))  __bf16 v8bf;
typedef __attribute__((ext_vector_type(4)))  __bf16 v4bf;
typedef __attribute__((ext_vector_type(8)))  float  v8f;

#define DIMC   1024
#define SEQ    2048
#define BATCH  4
#define HEADS  16
#define HDIM   64
#define ROWS   (BATCH * SEQ)     // 8192
#define NQKV   (3 * DIMC)        // 3072
#define ATTN_SCALE 0.125f        // 64^-0.5

#define LDK 72                   // padded LDS stride (bf16 elements), 144B rows

static __device__ __forceinline__ v8f wmma_bf16(v16bf a, v16bf b, v8f c) {
    // D = A(16x32) * B(32x16) + C(16x16 f32)
    return __builtin_amdgcn_wmma_f32_16x16x32_bf16(
        /*neg_a=*/false, a, /*neg_b=*/false, b,
        /*c_mod=*/(short)0, c, /*reuse_a=*/false, /*reuse_b=*/false);
}

// A fragment: 16(M) x 32(K) tile, row-major at p with row stride ld.
// ISA layout: lane = M (mod 16); lanes<16 hold K {0..7,16..23}, lanes>=16 hold
// K {8..15,24..31}; element f[j] = K kb+j (j<8), f[8+j] = K 16+kb+j.
static __device__ __forceinline__ v16bf frag_a(const __bf16* p, int ld) {
    const int lane = threadIdx.x & 31;
    const int r    = lane & 15;
    const int kb   = (lane >> 4) << 3;      // 0 or 8
    const __bf16* row = p + r * ld;
    v8bf lo = *(const v8bf*)(row + kb);
    v8bf hi = *(const v8bf*)(row + 16 + kb);
    return __builtin_shufflevector(lo, hi, 0,1,2,3,4,5,6,7,8,9,10,11,12,13,14,15);
}

// B fragment: 32(K) x 16(N) operand stored N-major: p[n*ld + k].
// ISA layout: lane = N (mod 16); lanes<16 hold K {0..7,16..23}, lanes>=16 hold K {8..15,24..31}.
static __device__ __forceinline__ v16bf frag_bt(const __bf16* p, int ld) {
    const int lane = threadIdx.x & 31;
    const int n    = lane & 15;
    const int kb   = (lane >> 4) << 3;      // 0 or 8
    const __bf16* row = p + n * ld + kb;
    v8bf lo = *(const v8bf*)(row);
    v8bf hi = *(const v8bf*)(row + 16);
    return __builtin_shufflevector(lo, hi, 0,1,2,3,4,5,6,7,8,9,10,11,12,13,14,15);
}

// ---------------------------------------------------------------------------
// Kernel 1: QKV projection.  C[8192,3072] = x[8192,1024] @ w[1024,3072] + b.
// 256 threads, 128x128 block tile, 8 waves each computing 32(M) x 64(N).
// Output scattered to q/k/v bf16 buffers laid out [b, h, n, d].
// ---------------------------------------------------------------------------
__global__ __launch_bounds__(256)
void qkv_gemm(const float* __restrict__ x, const float* __restrict__ wq,
              const float* __restrict__ bias,
              __bf16* __restrict__ qb, __bf16* __restrict__ kbuf,
              __bf16* __restrict__ vb) {
    __shared__ __align__(16) __bf16 As[128 * LDK];  // A tile, row-major MxK
    __shared__ __align__(16) __bf16 Bs[128 * LDK];  // B tile, N-major NxK

    const int t    = threadIdx.x;
    const int lane = t & 31;
    const int wid  = t >> 5;
    const int wm   = wid & 3;        // M group (32 rows)
    const int wn   = wid >> 2;       // N group (64 cols)
    const int rowBase = blockIdx.y * 128;
    const int colBase = blockIdx.x * 128;

    v8f acc[2][4] = {};

    for (int kt = 0; kt < DIMC; kt += 64) {
        // stage A: 128x64 fp32 -> bf16
        {
            const int r = t >> 1, c0 = (t & 1) * 32;
            const float* gp = x + (size_t)(rowBase + r) * DIMC + kt + c0;
            __bf16* sp = As + r * LDK + c0;
#pragma unroll
            for (int i = 0; i < 8; ++i) {
                float4 v = *(const float4*)(gp + 4 * i);
                v4bf b4 = { (__bf16)v.x, (__bf16)v.y, (__bf16)v.z, (__bf16)v.w };
                *(v4bf*)(sp + 4 * i) = b4;
            }
        }
        // stage B transposed: Bs[n*LDK + k], 64(k) x 128(n) source
        {
            const int kk = t >> 2, n0 = (t & 3) * 32;
            const float* gp = wq + (size_t)(kt + kk) * NQKV + colBase + n0;
#pragma unroll
            for (int i = 0; i < 8; ++i) {
                float4 v = *(const float4*)(gp + 4 * i);
                Bs[(n0 + 4 * i + 0) * LDK + kk] = (__bf16)v.x;
                Bs[(n0 + 4 * i + 1) * LDK + kk] = (__bf16)v.y;
                Bs[(n0 + 4 * i + 2) * LDK + kk] = (__bf16)v.z;
                Bs[(n0 + 4 * i + 3) * LDK + kk] = (__bf16)v.w;
            }
        }
        __syncthreads();
#pragma unroll
        for (int c = 0; c < 2; ++c) {
            v16bf a0 = frag_a(As + (wm * 32 + 0)  * LDK + c * 32, LDK);
            v16bf a1 = frag_a(As + (wm * 32 + 16) * LDK + c * 32, LDK);
            v16bf b0 = frag_bt(Bs + (wn * 64 + 0)  * LDK + c * 32, LDK);
            v16bf b1 = frag_bt(Bs + (wn * 64 + 16) * LDK + c * 32, LDK);
            v16bf b2 = frag_bt(Bs + (wn * 64 + 32) * LDK + c * 32, LDK);
            v16bf b3 = frag_bt(Bs + (wn * 64 + 48) * LDK + c * 32, LDK);
            acc[0][0] = wmma_bf16(a0, b0, acc[0][0]);
            acc[0][1] = wmma_bf16(a0, b1, acc[0][1]);
            acc[0][2] = wmma_bf16(a0, b2, acc[0][2]);
            acc[0][3] = wmma_bf16(a0, b3, acc[0][3]);
            acc[1][0] = wmma_bf16(a1, b0, acc[1][0]);
            acc[1][1] = wmma_bf16(a1, b1, acc[1][1]);
            acc[1][2] = wmma_bf16(a1, b2, acc[1][2]);
            acc[1][3] = wmma_bf16(a1, b3, acc[1][3]);
        }
        __syncthreads();
    }

    // epilogue: bias + scatter into q/k/v [b,h,n,d] (bf16)
    const int lc = lane & 15, lhalf = lane >> 4;
#pragma unroll
    for (int mi = 0; mi < 2; ++mi)
#pragma unroll
        for (int ni = 0; ni < 4; ++ni) {
            const int col = colBase + wn * 64 + ni * 16 + lc;
            const float bv = bias[col];
            const int which = col >> 10;       // 0=q 1=k 2=v
            const int rem = col & 1023;
            const int h = rem >> 6, d = rem & 63;
            __bf16* dst = (which == 0) ? qb : ((which == 1) ? kbuf : vb);
#pragma unroll
            for (int i = 0; i < 8; ++i) {
                const int m = rowBase + wm * 32 + mi * 16 + i + 8 * lhalf;
                const int bb = m >> 11;        // batch
                const int n  = m & 2047;       // seq pos
                const float val = acc[mi][ni][i] + bv;
                dst[(((size_t)(bb * HEADS + h)) * SEQ + n) * HDIM + d] = (__bf16)val;
            }
        }
}

// ---------------------------------------------------------------------------
// Kernel 2: flash attention. One block = one (b*h, 64-query block).
// 4 waves, each owns 16 query rows; online softmax over 64-key blocks.
// Output [b, n, h*64+d] bf16 for the projection GEMM.
// ---------------------------------------------------------------------------
__global__ __launch_bounds__(128)
void flash_attn(const __bf16* __restrict__ q, const __bf16* __restrict__ k,
                const __bf16* __restrict__ v, __bf16* __restrict__ o) {
    __shared__ __align__(16) __bf16 Qs[64 * LDK];   // [qrow][d]
    __shared__ __align__(16) __bf16 Ks[64 * LDK];   // [key][d]
    __shared__ __align__(16) __bf16 Vt[64 * LDK];   // [d][key] (transposed)
    __shared__ __align__(16) __bf16 Ps[64 * LDK];   // per-wave 16x64 P tiles

    const int t = threadIdx.x, lane = t & 31, wid = t >> 5;
    const int lc = lane & 15, lhalf = lane >> 4;
    const int bh = blockIdx.x;        // 0..63  (b*16 + h)
    const int qblk = blockIdx.y;      // 0..31

    const size_t base = (size_t)bh * SEQ * HDIM;
    const __bf16* qp = q + base;
    const __bf16* kp = k + base;
    const __bf16* vp = v + base;

    // stage Q (64 x 64 bf16)
    {
        const int r = t >> 1, c0 = (t & 1) * 32;
        const __bf16* gp = qp + (size_t)(qblk * 64 + r) * HDIM + c0;
        __bf16* sp = Qs + r * LDK + c0;
#pragma unroll
        for (int i = 0; i < 4; ++i)
            *(v8bf*)(sp + 8 * i) = *(const v8bf*)(gp + 8 * i);
    }
    __syncthreads();
    v16bf qf[2];
    qf[0] = frag_a(Qs + (wid * 16) * LDK + 0,  LDK);
    qf[1] = frag_a(Qs + (wid * 16) * LDK + 32, LDK);

    v8f oacc[4] = {};
    float mrow[8], lrow[8];
#pragma unroll
    for (int i = 0; i < 8; ++i) { mrow[i] = -1e30f; lrow[i] = 0.0f; }

    __bf16* Pw = Ps + wid * 16 * LDK;

    for (int kb = 0; kb < SEQ; kb += 64) {
        __syncthreads();   // previous iteration done with Ks/Vt
        // stage K [key][d]
        {
            const int r = t >> 1, c0 = (t & 1) * 32;
            const __bf16* gp = kp + (size_t)(kb + r) * HDIM + c0;
            __bf16* sp = Ks + r * LDK + c0;
#pragma unroll
            for (int i = 0; i < 4; ++i)
                *(v8bf*)(sp + 8 * i) = *(const v8bf*)(gp + 8 * i);
        }
        // stage V transposed: Vt[d][key]
        {
            const int r = t >> 1, c0 = (t & 1) * 32;
            const __bf16* gp = vp + (size_t)(kb + r) * HDIM + c0;
#pragma unroll
            for (int i = 0; i < 4; ++i) {
                v8bf x8 = *(const v8bf*)(gp + 8 * i);
#pragma unroll
                for (int j = 0; j < 8; ++j)
                    Vt[(c0 + 8 * i + j) * LDK + r] = x8[j];
            }
        }
        if (kb + 64 < SEQ) {  // global_prefetch_b8 for the next K/V blocks
            __builtin_prefetch(kp + (size_t)(kb + 64) * HDIM + lane * 8, 0, 0);
            __builtin_prefetch(vp + (size_t)(kb + 64) * HDIM + lane * 8, 0, 0);
        }
        __syncthreads();

        // S = Q @ K^T  (per wave: 16 rows x 64 keys, four 16x16 tiles)
        v8f s[4];
#pragma unroll
        for (int nt = 0; nt < 4; ++nt) {
            v8f z = {};
            z = wmma_bf16(qf[0], frag_bt(Ks + (nt * 16) * LDK + 0,  LDK), z);
            z = wmma_bf16(qf[1], frag_bt(Ks + (nt * 16) * LDK + 32, LDK), z);
            s[nt] = z;
        }

        // online softmax (row reductions across 16-lane halves)
#pragma unroll
        for (int i = 0; i < 8; ++i) {
            float tm = -1e30f;
#pragma unroll
            for (int nt = 0; nt < 4; ++nt) tm = fmaxf(tm, s[nt][i]);
            tm *= ATTN_SCALE;
#pragma unroll
            for (int msk = 1; msk < 16; msk <<= 1)
                tm = fmaxf(tm, __shfl_xor(tm, msk, 32));
            const float nm = fmaxf(mrow[i], tm);
            const float corr = __expf(mrow[i] - nm);
            mrow[i] = nm;
            lrow[i] *= corr;
#pragma unroll
            for (int nt = 0; nt < 4; ++nt) oacc[nt][i] *= corr;
            float psum = 0.0f;
#pragma unroll
            for (int nt = 0; nt < 4; ++nt) {
                const float pv = __expf(s[nt][i] * ATTN_SCALE - nm);
                s[nt][i] = pv;
                psum += pv;
            }
#pragma unroll
            for (int msk = 1; msk < 16; msk <<= 1)
                psum += __shfl_xor(psum, msk, 32);
            lrow[i] += psum;
        }

        // P (C-layout f32) -> per-wave LDS as bf16 A-layout tile.
        // Same-wave LDS ops are in-order: no barrier needed before reload.
#pragma unroll
        for (int nt = 0; nt < 4; ++nt)
#pragma unroll
            for (int i = 0; i < 8; ++i)
                Pw[(i + 8 * lhalf) * LDK + nt * 16 + lc] = (__bf16)s[nt][i];

        // O += P @ V
#pragma unroll
        for (int c = 0; c < 2; ++c) {
            v16bf pf = frag_a(Pw + c * 32, LDK);
#pragma unroll
            for (int nt2 = 0; nt2 < 4; ++nt2)
                oacc[nt2] = wmma_bf16(pf,
                    frag_bt(Vt + (nt2 * 16) * LDK + c * 32, LDK), oacc[nt2]);
        }
    }

    // epilogue: normalize by row sums, write [b, n, h*64+d] bf16
    const int bb = bh >> 4, hh = bh & 15;
#pragma unroll
    for (int nt2 = 0; nt2 < 4; ++nt2)
#pragma unroll
        for (int i = 0; i < 8; ++i) {
            const int n = qblk * 64 + wid * 16 + i + 8 * lhalf;
            const int d = nt2 * 16 + lc;
            const float val = oacc[nt2][i] / lrow[i];
            o[((size_t)(bb * SEQ + n)) * DIMC + hh * HDIM + d] = (__bf16)val;
        }
}

// ---------------------------------------------------------------------------
// Kernel 3: output projection.  out[8192,1024] = A(bf16) @ w_proj + b_proj.
// 256 threads, 128x128 block tile, 8 waves each computing 32(M) x 64(N).
// ---------------------------------------------------------------------------
__global__ __launch_bounds__(256)
void proj_gemm(const __bf16* __restrict__ a, const float* __restrict__ wp,
               const float* __restrict__ bias, float* __restrict__ out) {
    __shared__ __align__(16) __bf16 As[128 * LDK];
    __shared__ __align__(16) __bf16 Bs[128 * LDK];

    const int t = threadIdx.x, lane = t & 31, wid = t >> 5;
    const int wm = wid & 3;
    const int wn = wid >> 2;
    const int rowBase = blockIdx.y * 128;
    const int colBase = blockIdx.x * 128;

    v8f acc[2][4] = {};

    for (int kt = 0; kt < DIMC; kt += 64) {
        // stage A (already bf16): straight 16B copies, 128x64
        {
            const int r = t >> 1, c0 = (t & 1) * 32;
            const __bf16* gp = a + (size_t)(rowBase + r) * DIMC + kt + c0;
            __bf16* sp = As + r * LDK + c0;
#pragma unroll
            for (int i = 0; i < 4; ++i)
                *(v8bf*)(sp + 8 * i) = *(const v8bf*)(gp + 8 * i);
        }
        // stage B transposed: 64(k) x 128(n)
        {
            const int kk = t >> 2, n0 = (t & 3) * 32;
            const float* gp = wp + (size_t)(kt + kk) * DIMC + colBase + n0;
#pragma unroll
            for (int i = 0; i < 8; ++i) {
                float4 v = *(const float4*)(gp + 4 * i);
                Bs[(n0 + 4 * i + 0) * LDK + kk] = (__bf16)v.x;
                Bs[(n0 + 4 * i + 1) * LDK + kk] = (__bf16)v.y;
                Bs[(n0 + 4 * i + 2) * LDK + kk] = (__bf16)v.z;
                Bs[(n0 + 4 * i + 3) * LDK + kk] = (__bf16)v.w;
            }
        }
        __syncthreads();
#pragma unroll
        for (int c = 0; c < 2; ++c) {
            v16bf a0 = frag_a(As + (wm * 32 + 0)  * LDK + c * 32, LDK);
            v16bf a1 = frag_a(As + (wm * 32 + 16) * LDK + c * 32, LDK);
            v16bf b0 = frag_bt(Bs + (wn * 64 + 0)  * LDK + c * 32, LDK);
            v16bf b1 = frag_bt(Bs + (wn * 64 + 16) * LDK + c * 32, LDK);
            v16bf b2 = frag_bt(Bs + (wn * 64 + 32) * LDK + c * 32, LDK);
            v16bf b3 = frag_bt(Bs + (wn * 64 + 48) * LDK + c * 32, LDK);
            acc[0][0] = wmma_bf16(a0, b0, acc[0][0]);
            acc[0][1] = wmma_bf16(a0, b1, acc[0][1]);
            acc[0][2] = wmma_bf16(a0, b2, acc[0][2]);
            acc[0][3] = wmma_bf16(a0, b3, acc[0][3]);
            acc[1][0] = wmma_bf16(a1, b0, acc[1][0]);
            acc[1][1] = wmma_bf16(a1, b1, acc[1][1]);
            acc[1][2] = wmma_bf16(a1, b2, acc[1][2]);
            acc[1][3] = wmma_bf16(a1, b3, acc[1][3]);
        }
        __syncthreads();
    }

    const int lc = lane & 15, lhalf = lane >> 4;
#pragma unroll
    for (int mi = 0; mi < 2; ++mi)
#pragma unroll
        for (int ni = 0; ni < 4; ++ni) {
            const int col = colBase + wn * 64 + ni * 16 + lc;
            const float bv = bias[col];
#pragma unroll
            for (int i = 0; i < 8; ++i) {
                const int m = rowBase + wm * 32 + mi * 16 + i + 8 * lhalf;
                out[(size_t)m * DIMC + col] = acc[mi][ni][i] + bv;
            }
        }
}

// ---------------------------------------------------------------------------
extern "C" void kernel_launch(void* const* d_in, const int* in_sizes, int n_in,
                              void* d_out, int out_size, void* d_ws, size_t ws_size,
                              hipStream_t stream) {
    const float* x      = (const float*)d_in[0];
    const float* w_qkv  = (const float*)d_in[1];
    const float* b_qkv  = (const float*)d_in[2];
    const float* w_proj = (const float*)d_in[3];
    const float* b_proj = (const float*)d_in[4];
    float* out = (float*)d_out;

    const size_t qkv_elems = (size_t)ROWS * DIMC;   // 8M elements each
    __bf16* qb = (__bf16*)d_ws;                     // 16 MB
    __bf16* kb = qb + qkv_elems;                    // 16 MB
    __bf16* vb = kb + qkv_elems;                    // 16 MB
    __bf16* ao = vb + qkv_elems;                    // 16 MB (attn output)

    qkv_gemm<<<dim3(NQKV / 128, ROWS / 128), 256, 0, stream>>>(
        x, w_qkv, b_qkv, qb, kb, vb);
    flash_attn<<<dim3(BATCH * HEADS, SEQ / 64), 128, 0, stream>>>(
        qb, kb, vb, ao);
    proj_gemm<<<dim3(DIMC / 128, ROWS / 128), 256, 0, stream>>>(
        ao, w_proj, b_proj, out);
}
